// get_model_56538949484643
// MI455X (gfx1250) — matrix operand order repaired and checked by reading, not compiled, not run on hardware
//
#include <hip/hip_runtime.h>
#include <hip/hip_bf16.h>

#define EPS   1e-5f
#define NPTS  2048
#define KNN   20
#define BATCH 4

typedef __attribute__((ext_vector_type(16))) __bf16 v16bf;
typedef __attribute__((ext_vector_type(8)))  float  v8f;
typedef __attribute__((ext_vector_type(2)))  float  v2f;

__device__ __forceinline__ unsigned short f2bf_bits(float x) {
    union { float f; unsigned u; } v; v.f = x;
    unsigned r = v.u + 0x7FFFu + ((v.u >> 16) & 1u);   // round-to-nearest-even
    return (unsigned short)(r >> 16);
}
__device__ __forceinline__ __bf16 bits2bf(unsigned short s) {
    return __builtin_bit_cast(__bf16, s);
}

// ---- CDNA5 async global->LDS copies (ASYNCcnt-tracked, no dest VGPRs) ----
__device__ __forceinline__ void async_copy_b128(unsigned lds, unsigned long long ga) {
    asm volatile("global_load_async_to_lds_b128 %0, %1, off" :: "v"(lds), "v"(ga) : "memory");
}
__device__ __forceinline__ void async_copy_b32(unsigned lds, unsigned long long ga) {
    asm volatile("global_load_async_to_lds_b32 %0, %1, off" :: "v"(lds), "v"(ga) : "memory");
}
template<int N>
__device__ __forceinline__ void wait_async_le() {      // ASYNCcnt <= N
    asm volatile("s_wait_asynccnt %0" :: "n"(N) : "memory");
}
__device__ __forceinline__ unsigned lds_off(const void* p) {
    return (unsigned)(uintptr_t)p;      // LDS aperture: addr[31:0] = LDS byte offset
}

// ---------------------------------------------------------------- BN prep
__global__ void bn_prep_kernel(const float* g, const float* b, const float* m,
                               const float* v, float* sc, float* bi, int C) {
    int t = blockIdx.x * blockDim.x + threadIdx.x;
    if (t < C) {
        float s = g[t] * rsqrtf(v[t] + EPS);
        sc[t] = s;
        bi[t] = b[t] - m[t] * s;
    }
}

// ------------------------------------------------- weight -> bf16 (padded)
__global__ void convert_w_kernel(const float* w, unsigned short* wb,
                                 int O, int Cin, int Cp) {
    int t = blockIdx.x * blockDim.x + threadIdx.x;
    if (t < O * Cp) {
        int o = t / Cp, c = t % Cp;
        float val = (c < Cin) ? w[o * Cin + c] : 0.f;
        wb[t] = f2bf_bits(val);
    }
}

// ----------------------------------------------------- point norms ||x||^2
__global__ void norms_kernel(const float* act, float* norms, int CIN, int total) {
    int t = blockIdx.x * blockDim.x + threadIdx.x;
    if (t < total) {
        const float* r = act + (size_t)t * CIN;
        float s = 0.f;
        for (int c = 0; c < CIN; ++c) s += r[c] * r[c];
        norms[t] = s;
    }
}

// ------------------------------------------- kNN via fp32 WMMA Gram matrix
// one wave per 16-query tile; pd = 2*X Xc^T - ||q||^2 - ||c||^2
template<int CIN>
__global__ void knn_wmma_kernel(const float* __restrict__ act,
                                const float* __restrict__ norms,
                                int* __restrict__ idx) {
    constexpr int NCH = (CIN + 3) / 4;
    __shared__ float s_qq[16];
    __shared__ float s_pd[16][17];
    int b  = blockIdx.y;
    int n0 = blockIdx.x * 16;
    int lane = threadIdx.x;                 // blockDim = 32
    const size_t bN = (size_t)b * NPTS;
    const float* base = act + bN * CIN;

    int mq     = lane & 15;
    int kbase  = (lane & 16) ? 2 : 0;
    int rowoff = (lane & 16) ? 8 : 0;

    v2f afr[NCH];                            // query fragments, registers
#pragma unroll
    for (int cc = 0; cc < NCH; ++cc) {
        int c0 = cc * 4 + kbase;
        afr[cc].x = (c0     < CIN) ? base[(size_t)(n0 + mq) * CIN + c0]     : 0.f;
        afr[cc].y = (c0 + 1 < CIN) ? base[(size_t)(n0 + mq) * CIN + c0 + 1] : 0.f;
    }
    if (lane < 16) s_qq[lane] = norms[bN + n0 + lane];
    __syncthreads();

    float best[KNN]; int bidx[KNN];
#pragma unroll
    for (int j = 0; j < KNN; ++j) { best[j] = -__builtin_inff(); bidx[j] = 0; }

#pragma clang loop unroll(disable)
    for (int t = 0; t < NPTS / 16; ++t) {
        int cand = t * 16 + (lane & 15);
        if (t + 1 < NPTS / 16)
            __builtin_prefetch(&base[(size_t)(cand + 16) * CIN], 0, 0);
        v8f acc = {};
#pragma unroll
        for (int cc = 0; cc < NCH; ++cc) {
            int c0 = cc * 4 + kbase;
            v2f bfr;
            bfr.x = (c0     < CIN) ? base[(size_t)cand * CIN + c0]     : 0.f;
            bfr.y = (c0 + 1 < CIN) ? base[(size_t)cand * CIN + c0 + 1] : 0.f;
            acc = __builtin_amdgcn_wmma_f32_16x16x4_f32(false, afr[cc], false, bfr,
                                                        (short)0, acc, false, false);
        }
        float nn = norms[bN + cand];
#pragma unroll
        for (int i = 0; i < 8; ++i)
            s_pd[rowoff + i][lane & 15] = 2.f * acc[i] - s_qq[rowoff + i] - nn;
        __syncthreads();
        if (lane < 16) {
            // outer scan rolled (code size); insertion network stays unrolled
#pragma clang loop unroll(disable)
            for (int j2 = 0; j2 < 16; ++j2) {
                float pd = s_pd[lane][j2];
                if (pd > best[KNN - 1]) {
                    float v = pd; int vi = t * 16 + j2;
#pragma unroll
                    for (int j = 0; j < KNN; ++j) {
                        if (v > best[j]) {
                            float tv = best[j]; int ti = bidx[j];
                            best[j] = v; bidx[j] = vi; v = tv; vi = ti;
                        }
                    }
                }
            }
        }
        __syncthreads();
    }
    if (lane < 16) {
        int* op = idx + (bN + n0 + lane) * KNN;
#pragma unroll
        for (int j = 0; j < KNN; ++j) op[j] = bidx[j];
    }
}

// ---- neighbor-row gather: async global->LDS, one k slice ----
template<int CIN>
__device__ __forceinline__ void issue_gather(const float* __restrict__ act_in, size_t bN,
                                             const int (*s_idx)[KNN],
                                             float (*dst)[CIN], int k, int tid) {
    if constexpr (CIN % 4 == 0) {
        constexpr int CH = CIN / 4;
        for (int t = tid; t < 16 * CH; t += 128) {
            int r = t / CH, c4 = t % CH;
            int nb = s_idx[r][k];
            async_copy_b128(lds_off(&dst[r][c4 * 4]),
                            (unsigned long long)(uintptr_t)&act_in[(bN + nb) * CIN + c4 * 4]);
        }
    } else {
        for (int t = tid; t < 16 * CIN; t += 128) {
            int r = t / CIN, c = t % CIN;
            int nb = s_idx[r][k];
            async_copy_b32(lds_off(&dst[r][c]),
                           (unsigned long long)(uintptr_t)&act_in[(bN + nb) * CIN + c]);
        }
    }
}

// ------------------------------------------- fused EdgeConv (WMMA bf16)
// block = 128 threads (4 waves); chunk-outer loop (A reused OTW times);
// double-buffered async gather overlaps k+1 copies with k's WMMA work.
template<int CIN, int COUT>
__global__ void edgeconv_kernel(const float* __restrict__ act_in,
                                const int*   __restrict__ idx,
                                const unsigned short* __restrict__ wb,
                                const float* __restrict__ scale,
                                const float* __restrict__ bias,
                                float* __restrict__ act_out) {
    constexpr int C2  = 2 * CIN;
    constexpr int C2P = (C2 < 32) ? 32 : C2;
    constexpr int NCHUNK = C2P / 32;
    constexpr int NOT = COUT / 16;
    constexpr int WAVES = 4;
    constexpr int OTW = NOT / WAVES;
    // per-wave async instructions per gather slice
    constexpr int ISS = (CIN % 4 == 0)
        ? ((16 * (CIN / 4)) / 128 > 0 ? (16 * (CIN / 4)) / 128 : 1) : 1;

    __shared__ float          s_ctrf[16][CIN];
    __shared__ float          s_nbrf[2][16][CIN];
    __shared__ unsigned short s_ctr [16][CIN];
    __shared__ unsigned short s_edge[16][CIN];
    __shared__ int            s_idx [16][KNN];

    int b   = blockIdx.y;
    int n0  = blockIdx.x * 16;
    int tid = threadIdx.x;
    const size_t bN = (size_t)b * NPTS;

    // centers: contiguous 16*CIN floats -> async b128 into LDS
    {
        const float* cbase = &act_in[(bN + n0) * CIN];
        for (int t = tid; t < (16 * CIN) / 4; t += 128)
            async_copy_b128(lds_off(&s_ctrf[0][0]) + t * 16,
                            (unsigned long long)(uintptr_t)cbase + t * 16);
    }
    for (int t = tid; t < 16 * KNN; t += 128) {
        int m = t / KNN, k = t % KNN;
        s_idx[m][k] = idx[(bN + n0 + m) * KNN + k];
    }
    wait_async_le<0>();
    __syncthreads();
    for (int t = tid; t < 16 * CIN; t += 128)
        ((unsigned short*)s_ctr)[t] = f2bf_bits(((const float*)s_ctrf)[t]);
    __syncthreads();

    int wave = tid >> 5, lane = tid & 31;
    int mrow = lane & 15;
    int col  = lane & 15;
    int kadd = (lane & 16) ? 16 : 0;
    int kboff = (lane & 16) ? 16 : 0;

    float vmax[OTW][8];
#pragma unroll
    for (int j = 0; j < OTW; ++j)
#pragma unroll
        for (int i = 0; i < 8; ++i) vmax[j][i] = -__builtin_inff();

    issue_gather<CIN>(act_in, bN, s_idx, s_nbrf[0], 0, tid);

#pragma clang loop unroll(disable)
    for (int k = 0; k < KNN; ++k) {
        int cur = k & 1;
        if (k + 1 < KNN) {
            issue_gather<CIN>(act_in, bN, s_idx, s_nbrf[cur ^ 1], k + 1, tid);
            wait_async_le<ISS>();     // cur slice done; k+1 slice may be in flight
        } else {
            wait_async_le<0>();
        }
        __syncthreads();
        for (int t = tid; t < 16 * CIN; t += 128) {
            int m = t / CIN, c = t % CIN;
            s_edge[m][c] = f2bf_bits(s_nbrf[cur][m][c] - s_ctrf[m][c]);
        }
        __syncthreads();

        v8f acc[OTW];
#pragma unroll
        for (int j = 0; j < OTW; ++j) acc[j] = (v8f){};
#pragma unroll
        for (int cc = 0; cc < NCHUNK; ++cc) {
            v16bf a;                              // built ONCE per chunk
#pragma unroll
            for (int e = 0; e < 16; ++e) {
                int kl = kadd + e;
                int cg = cc * 32 + kl;
                unsigned short bits = 0;
                if (cg < CIN)      bits = s_edge[mrow][cg];
                else if (cg < C2)  bits = s_ctr[mrow][cg - CIN];
                a[e] = bits2bf(bits);
            }
            int kb = cc * 32 + kboff;
#pragma unroll
            for (int j = 0; j < OTW; ++j) {
                int ot = wave * OTW + j;
                v16bf bm = *(const v16bf*)(const void*)(wb + (size_t)(ot * 16 + col) * C2P + kb);
                acc[j] = __builtin_amdgcn_wmma_f32_16x16x32_bf16(false, a, false, bm,
                                                                 (short)0, acc[j], false, false);
            }
        }
#pragma unroll
        for (int j = 0; j < OTW; ++j) {
            int o = (wave * OTW + j) * 16 + col;
            float s = scale[o], bb = bias[o];
#pragma unroll
            for (int i = 0; i < 8; ++i) {
                float y = acc[j][i] * s + bb;
                y = (y >= 0.f) ? y : 0.2f * y;
                vmax[j][i] = fmaxf(vmax[j][i], y);
            }
        }
        __syncthreads();
    }
    int rowoff = (lane & 16) ? 8 : 0;
#pragma unroll
    for (int j = 0; j < OTW; ++j) {
        int o = (wave * OTW + j) * 16 + col;
#pragma unroll
        for (int i = 0; i < 8; ++i) {
            int n = n0 + rowoff + i;
            act_out[(bN + n) * COUT + o] = vmax[j][i];
        }
    }
}

// ------------------------------------------------ concat x1..x4 -> bf16
__global__ void cat_kernel(const float* a1, const float* a2, const float* a3,
                           const float* a4, unsigned short* catb) {
    size_t t = (size_t)blockIdx.x * blockDim.x + threadIdx.x;
    size_t total = (size_t)BATCH * NPTS * 512;
    if (t >= total) return;
    int c = (int)(t % 512);
    size_t bn = t / 512;
    float v;
    if (c < 64)       v = a1[bn * 64 + c];
    else if (c < 128) v = a2[bn * 64 + (c - 64)];
    else if (c < 256) v = a3[bn * 128 + (c - 128)];
    else              v = a4[bn * 256 + (c - 256)];
    catb[t] = f2bf_bits(v);
}

// ------------------------------ w5 conv (WMMA) + fused partial max/sum
// block = 512 threads (16 waves); A staged via async; A chunk reused 4x.
__global__ void hconv_kernel(const unsigned short* __restrict__ catb,
                             const unsigned short* __restrict__ wb,
                             const float* __restrict__ scale,
                             const float* __restrict__ bias,
                             float* __restrict__ pmax, float* __restrict__ psum) {
    __shared__ unsigned short s_cat[16][512];
    int b = blockIdx.y;
    int tile = blockIdx.x;
    int n0 = tile * 16;
    int tid = threadIdx.x;
    int wave = tid >> 5, lane = tid & 31;
    int mrow = lane & 15, col = lane & 15;
    int kadd = (lane & 16) ? 16 : 0;
    int kboff = (lane & 16) ? 16 : 0;
    const size_t bN = (size_t)b * NPTS;

    // stage contiguous 16x512 bf16 tile (16KB) through async b128
    {
        const unsigned short* gsrc = catb + (bN + n0) * 512;
        for (int t = tid; t < 1024; t += 512)
            async_copy_b128(lds_off(&s_cat[0][0]) + t * 16,
                            (unsigned long long)(uintptr_t)gsrc + t * 16);
    }
    wait_async_le<0>();
    __syncthreads();

    const unsigned short* arow = &s_cat[mrow][0];
    v8f acc[4];
#pragma unroll
    for (int j = 0; j < 4; ++j) acc[j] = (v8f){};
#pragma unroll
    for (int cc = 0; cc < 16; ++cc) {
        v16bf a;                                  // built ONCE per chunk
#pragma unroll
        for (int e = 0; e < 16; ++e) {
            int cg = cc * 32 + kadd + e;
            a[e] = bits2bf(arow[cg]);
        }
        int kb = cc * 32 + kboff;
#pragma unroll
        for (int j = 0; j < 4; ++j) {
            int ot = wave * 4 + j;
            v16bf bm = *(const v16bf*)(const void*)(wb + (size_t)(ot * 16 + col) * 512 + kb);
            acc[j] = __builtin_amdgcn_wmma_f32_16x16x32_bf16(false, a, false, bm,
                                                             (short)0, acc[j], false, false);
        }
    }
#pragma unroll
    for (int j = 0; j < 4; ++j) {
        int o = (wave * 4 + j) * 16 + col;
        float s = scale[o], bb = bias[o];
        float mx = -__builtin_inff(), sm = 0.f;
#pragma unroll
        for (int i = 0; i < 8; ++i) {
            float y = acc[j][i] * s + bb;
            y = (y >= 0.f) ? y : 0.2f * y;
            mx = fmaxf(mx, y); sm += y;
        }
        mx = fmaxf(mx, __shfl_xor(mx, 16, 32));
        sm += __shfl_xor(sm, 16, 32);
        if (lane < 16) {
            size_t po = ((size_t)b * 128 + tile) * 1024 + o;
            pmax[po] = mx;
            psum[po] = sm;
        }
    }
}

// ---------------------------------------------- global max/mean reduce
__global__ void reduce_z_kernel(const float* pmax, const float* psum,
                                float* z, float* mid1) {
    int t = blockIdx.x * blockDim.x + threadIdx.x;
    if (t >= BATCH * 1024) return;
    int b = t / 1024, o = t % 1024;
    float mx = -__builtin_inff(), sm = 0.f;
#pragma unroll 4
    for (int i = 0; i < 128; ++i) {
        size_t po = ((size_t)b * 128 + i) * 1024 + o;
        mx = fmaxf(mx, pmax[po]);
        sm += psum[po];
    }
    float mean = sm * (1.f / NPTS);
    z[b * 2048 + o] = mx;
    z[b * 2048 + 1024 + o] = mean;
    mid1[b * 2048 + o] = mx;
    mid1[b * 2048 + 1024 + o] = mean;
}

// ---------------------------------------------------------- MLP head
__global__ void fc1_kernel(const float* z, const float* lw1,
                           const float* g, const float* bb, const float* m, const float* v,
                           float* z2, float* mid2) {
    int t = blockIdx.x * blockDim.x + threadIdx.x;
    if (t >= BATCH * 512) return;
    int b = t / 512, o = t % 512;
    const float* zr = z + b * 2048;
    const float* wr = lw1 + (size_t)o * 2048;
    float acc = 0.f;
#pragma unroll 8
    for (int c = 0; c < 2048; ++c) acc += zr[c] * wr[c];
    float s = g[o] * rsqrtf(v[o] + EPS);
    float y = (acc - m[o]) * s + bb[o];
    y = (y >= 0.f) ? y : 0.2f * y;
    z2[t] = y; mid2[t] = y;
}

__global__ void fc2_kernel(const float* z2, const float* lw2, const float* lb2,
                           const float* g, const float* bb, const float* m, const float* v,
                           float* z3, float* mid3) {
    int t = blockIdx.x * blockDim.x + threadIdx.x;
    if (t >= BATCH * 256) return;
    int b = t / 256, o = t % 256;
    const float* zr = z2 + b * 512;
    const float* wr = lw2 + (size_t)o * 512;
    float acc = lb2[o];
#pragma unroll 8
    for (int c = 0; c < 512; ++c) acc += zr[c] * wr[c];
    float s = g[o] * rsqrtf(v[o] + EPS);
    float y = (acc - m[o]) * s + bb[o];
    y = (y >= 0.f) ? y : 0.2f * y;
    z3[t] = y; mid3[t] = y;
}

__global__ void fc3_kernel(const float* z3, const float* lw3, const float* lb3,
                           float* out) {
    int b = blockIdx.x;
    int t = threadIdx.x;           // 64 threads
    __shared__ float sl[40];
    if (t < 40) {
        const float* zr = z3 + b * 256;
        const float* wr = lw3 + (size_t)t * 256;
        float acc = lb3[t];
        for (int c = 0; c < 256; ++c) acc += zr[c] * wr[c];
        sl[t] = acc;
    }
    __syncthreads();
    if (t < 40) {
        float mx = -__builtin_inff();
        for (int i = 0; i < 40; ++i) mx = fmaxf(mx, sl[i]);
        float lse = 0.f;
        for (int i = 0; i < 40; ++i) lse += __expf(sl[i] - mx);
        lse = __logf(lse);
        out[b * 40 + t] = sl[t] - mx - lse;
    }
}

// ================================================================ launch
extern "C" void kernel_launch(void* const* d_in, const int* in_sizes, int n_in,
                              void* d_out, int out_size, void* d_ws, size_t ws_size,
                              hipStream_t stream) {
    (void)in_sizes; (void)n_in; (void)out_size; (void)ws_size;
    const float* x   = (const float*)d_in[0];
    const float* w1  = (const float*)d_in[1];
    const float* g1  = (const float*)d_in[2];
    const float* b1  = (const float*)d_in[3];
    const float* m1  = (const float*)d_in[4];
    const float* v1  = (const float*)d_in[5];
    const float* w2  = (const float*)d_in[6];
    const float* g2  = (const float*)d_in[7];
    const float* b2  = (const float*)d_in[8];
    const float* m2  = (const float*)d_in[9];
    const float* v2  = (const float*)d_in[10];
    const float* w3  = (const float*)d_in[11];
    const float* g3  = (const float*)d_in[12];
    const float* b3  = (const float*)d_in[13];
    const float* m3  = (const float*)d_in[14];
    const float* v3  = (const float*)d_in[15];
    const float* w4  = (const float*)d_in[16];
    const float* g4  = (const float*)d_in[17];
    const float* b4  = (const float*)d_in[18];
    const float* m4  = (const float*)d_in[19];
    const float* v4  = (const float*)d_in[20];
    const float* w5  = (const float*)d_in[21];
    const float* g5  = (const float*)d_in[22];
    const float* b5  = (const float*)d_in[23];
    const float* m5  = (const float*)d_in[24];
    const float* v5  = (const float*)d_in[25];
    const float* lw1 = (const float*)d_in[26];
    const float* g6  = (const float*)d_in[27];
    const float* b6  = (const float*)d_in[28];
    const float* m6  = (const float*)d_in[29];
    const float* v6  = (const float*)d_in[30];
    const float* lw2 = (const float*)d_in[31];
    const float* lb2 = (const float*)d_in[32];
    const float* g7  = (const float*)d_in[33];
    const float* b7  = (const float*)d_in[34];
    const float* m7  = (const float*)d_in[35];
    const float* v7  = (const float*)d_in[36];
    const float* lw3 = (const float*)d_in[37];
    const float* lb3 = (const float*)d_in[38];

    float* outp = (float*)d_out;
    float* mid1 = outp + BATCH * 40;                 // 160
    float* mid2 = mid1 + BATCH * 2048;               // 8352
    float* mid3 = mid2 + BATCH * 512;                // 10400

    char* wsp = (char*)d_ws;
    size_t off = 0;
    auto carve = [&](size_t bytes) -> void* {
        void* p = wsp + off;
        off += (bytes + 255) & ~(size_t)255;
        return p;
    };
    int*   idxb  = (int*)  carve((size_t)BATCH * NPTS * KNN * 4);
    float* normsb= (float*)carve((size_t)BATCH * NPTS * 4);
    float* act1  = (float*)carve((size_t)BATCH * NPTS * 64  * 4);
    float* act2  = (float*)carve((size_t)BATCH * NPTS * 64  * 4);
    float* act3  = (float*)carve((size_t)BATCH * NPTS * 128 * 4);
    float* act4  = (float*)carve((size_t)BATCH * NPTS * 256 * 4);
    unsigned short* catb = (unsigned short*)carve((size_t)BATCH * NPTS * 512 * 2);
    unsigned short* wb1 = (unsigned short*)carve(64   * 32  * 2);
    unsigned short* wb2 = (unsigned short*)carve(64   * 128 * 2);
    unsigned short* wb3 = (unsigned short*)carve(128  * 128 * 2);
    unsigned short* wb4 = (unsigned short*)carve(256  * 256 * 2);
    unsigned short* wb5 = (unsigned short*)carve(1024 * 512 * 2);
    float* sc1 = (float*)carve(64 * 4);   float* bi1 = (float*)carve(64 * 4);
    float* sc2 = (float*)carve(64 * 4);   float* bi2 = (float*)carve(64 * 4);
    float* sc3 = (float*)carve(128 * 4);  float* bi3 = (float*)carve(128 * 4);
    float* sc4 = (float*)carve(256 * 4);  float* bi4 = (float*)carve(256 * 4);
    float* sc5 = (float*)carve(1024 * 4); float* bi5 = (float*)carve(1024 * 4);
    float* pmax = (float*)carve((size_t)BATCH * 128 * 1024 * 4);
    float* psum = (float*)carve((size_t)BATCH * 128 * 1024 * 4);
    float* zbuf = (float*)carve((size_t)BATCH * 2048 * 4);
    float* z2   = (float*)carve((size_t)BATCH * 512 * 4);
    float* z3   = (float*)carve((size_t)BATCH * 256 * 4);

    // parameter prep
    bn_prep_kernel<<<1, 64, 0, stream>>>(g1, b1, m1, v1, sc1, bi1, 64);
    bn_prep_kernel<<<1, 64, 0, stream>>>(g2, b2, m2, v2, sc2, bi2, 64);
    bn_prep_kernel<<<1, 128, 0, stream>>>(g3, b3, m3, v3, sc3, bi3, 128);
    bn_prep_kernel<<<1, 256, 0, stream>>>(g4, b4, m4, v4, sc4, bi4, 256);
    bn_prep_kernel<<<4, 256, 0, stream>>>(g5, b5, m5, v5, sc5, bi5, 1024);
    convert_w_kernel<<<(64 * 32 + 255) / 256, 256, 0, stream>>>(w1, wb1, 64, 6, 32);
    convert_w_kernel<<<(64 * 128 + 255) / 256, 256, 0, stream>>>(w2, wb2, 64, 128, 128);
    convert_w_kernel<<<(128 * 128 + 255) / 256, 256, 0, stream>>>(w3, wb3, 128, 128, 128);
    convert_w_kernel<<<(256 * 256 + 255) / 256, 256, 0, stream>>>(w4, wb4, 256, 256, 256);
    convert_w_kernel<<<(1024 * 512 + 255) / 256, 256, 0, stream>>>(w5, wb5, 1024, 512, 512);

    const int NPT = BATCH * NPTS;
    dim3 kg(NPTS / 16, BATCH);
    dim3 eg(NPTS / 16, BATCH);

    norms_kernel<<<(NPT + 255) / 256, 256, 0, stream>>>(x, normsb, 3, NPT);
    knn_wmma_kernel<3><<<kg, 32, 0, stream>>>(x, normsb, idxb);
    edgeconv_kernel<3, 64><<<eg, 128, 0, stream>>>(x, idxb, wb1, sc1, bi1, act1);

    norms_kernel<<<(NPT + 255) / 256, 256, 0, stream>>>(act1, normsb, 64, NPT);
    knn_wmma_kernel<64><<<kg, 32, 0, stream>>>(act1, normsb, idxb);
    edgeconv_kernel<64, 64><<<eg, 128, 0, stream>>>(act1, idxb, wb2, sc2, bi2, act2);

    norms_kernel<<<(NPT + 255) / 256, 256, 0, stream>>>(act2, normsb, 64, NPT);
    knn_wmma_kernel<64><<<kg, 32, 0, stream>>>(act2, normsb, idxb);
    edgeconv_kernel<64, 128><<<eg, 128, 0, stream>>>(act2, idxb, wb3, sc3, bi3, act3);

    norms_kernel<<<(NPT + 255) / 256, 256, 0, stream>>>(act3, normsb, 128, NPT);
    knn_wmma_kernel<128><<<kg, 32, 0, stream>>>(act3, normsb, idxb);
    edgeconv_kernel<128, 256><<<eg, 128, 0, stream>>>(act3, idxb, wb4, sc4, bi4, act4);

    size_t cat_total = (size_t)BATCH * NPTS * 512;
    cat_kernel<<<(unsigned)((cat_total + 255) / 256), 256, 0, stream>>>(act1, act2, act3, act4, catb);

    dim3 hg(NPTS / 16, BATCH);
    hconv_kernel<<<hg, 512, 0, stream>>>(catb, wb5, sc5, bi5, pmax, psum);
    reduce_z_kernel<<<(BATCH * 1024 + 255) / 256, 256, 0, stream>>>(pmax, psum, zbuf, mid1);

    fc1_kernel<<<(BATCH * 512 + 255) / 256, 256, 0, stream>>>(zbuf, lw1, g6, b6, m6, v6, z2, mid2);
    fc2_kernel<<<(BATCH * 256 + 255) / 256, 256, 0, stream>>>(z2, lw2, lb2, g7, b7, m7, v7, z3, mid3);
    fc3_kernel<<<BATCH, 64, 0, stream>>>(z3, lw3, lb3, outp);
}
